// MANO_SMPL_40570261078081
// MI455X (gfx1250) — compile-verified, matrix operand order checked by reading, not run on hardware
//
#include <hip/hip_runtime.h>

// MANO/SMPL LBS pipeline for MI455X (gfx1250, wave32).
//   k_jt      : Jt[21,3] = J_regressor^T . v_template            (tiny)
//   k_build_G : G[v][k=(j*4+q)] = weights[v][j]*vh[v][q]  (f16, batch-const)
//   k_pose    : per-batch euler->Rodrigues->FK chain -> A_p[b][64] (f16, p=0..2)
//   k_lbs     : WMMA GEMM  Verts_p(4096x1084) = A_p(4096x64) x G^T(64x1084)
//               one wave per (btile,vtile); p=0..2 fused -> 6 chained WMMAs,
//               B operand loaded once, stores contiguous 12B per (b,v).
//   k_joints  : joints[b,j,c] = sum_v verts[b,v,c]*Jreg[v,j]
//
// Workspace layout (needs ~1.7 MB):
//   [0,256)                        : Jt  (f32, 64 floats)
//   [256, 256+NVP*64*2)            : G   (f16, NVP x 64, padded rows zeroed)
//   [256+NVP*64*2, +3*B*64*2)      : A_p (f16, [3][B][64])

#define BB     4096
#define NV     1084
#define NVP    1088          // 68 * 16
#define NVT    68
#define NBT    256           // 4096/16
#define NJR    21
#define NC     45

typedef __attribute__((ext_vector_type(16))) _Float16 v16h;
typedef __attribute__((ext_vector_type(8)))  _Float16 v8h;
typedef __attribute__((ext_vector_type(8)))  float    v8f;

// ---------------------------------------------------------------- helpers
__device__ inline void rodr(float x, float y, float z, float* R) {
    float a2  = x*x + y*y + z*z + 1e-8f;
    float ang = sqrtf(a2);
    float inv = 1.0f / ang;
    float ax = x*inv, ay = y*inv, az = z*inv;
    float c = cosf(ang), s = sinf(ang), ic = 1.0f - c;
    R[0] = c + ic*ax*ax;      R[1] = ic*ax*ay - s*az;  R[2] = ic*ax*az + s*ay;
    R[3] = ic*ax*ay + s*az;   R[4] = c + ic*ay*ay;     R[5] = ic*ay*az - s*ax;
    R[6] = ic*ax*az - s*ay;   R[7] = ic*ay*az + s*ax;  R[8] = c + ic*az*az;
}

__device__ inline v16h load16(const _Float16* p) {
    // halves {0..7} and {16..23} relative to p (ISA 16-bit operand striping)
    union { v16h v; v8h h[2]; } u;
    u.h[0] = *reinterpret_cast<const v8h*>(p);
    u.h[1] = *reinterpret_cast<const v8h*>(p + 16);
    return u.v;
}

// ---------------------------------------------------------------- kernel 1a
__global__ void k_jt(const float* __restrict__ vtem,
                     const float* __restrict__ Jreg,
                     float* __restrict__ Jt) {
    int t = threadIdx.x;
    if (t >= 63) return;
    int j = t / 3, c = t % 3;
    float acc = 0.0f;
    for (int v = 0; v < NV; ++v)
        acc += vtem[v*3 + c] * Jreg[v*NJR + j];
    Jt[t] = acc;
}

// ---------------------------------------------------------------- kernel 1b
__global__ void k_build_G(const float* __restrict__ wgt,
                          const float* __restrict__ vtem,
                          unsigned short* __restrict__ G_u) {
    _Float16* G = reinterpret_cast<_Float16*>(G_u);
    int idx = blockIdx.x * blockDim.x + threadIdx.x;   // < NVP*64
    int v = idx >> 6, k = idx & 63;
    int j = k >> 2,  q = k & 3;
    float val = 0.0f;
    if (v < NV) {
        float vh = (q < 3) ? vtem[v*3 + q] : 1.0f;
        val = wgt[v*16 + j] * vh;
    }
    G[idx] = (_Float16)val;
}

// ---------------------------------------------------------------- kernel 2
__global__ void k_pose(const float* __restrict__ theta,
                       const float* __restrict__ wrist,
                       const float* __restrict__ hcomp,
                       const float* __restrict__ hmean,
                       const float* __restrict__ Jt,
                       unsigned short* __restrict__ Ah_u) {
    _Float16* Ah = reinterpret_cast<_Float16*>(Ah_u);
    int b = blockIdx.x * blockDim.x + threadIdx.x;
    if (b >= BB) return;

    float th[NC];
    for (int t = 0; t < NC; ++t) th[t] = theta[b*NC + t];

    const int par[16] = {-1,0,1,2,0,4,5,0,7,8,0,10,11,0,13,14};
    float Aw[16][12];   // [joint][3x4 row-major], bottom row implied 0001
    float R[9];

    // joint 0: global wrist rotation, absolute translation Jt[0]
    rodr(wrist[b*3+0], wrist[b*3+1], wrist[b*3+2], R);
    for (int p = 0; p < 3; ++p) {
        Aw[0][p*4+0] = R[p*3+0];
        Aw[0][p*4+1] = R[p*3+1];
        Aw[0][p*4+2] = R[p*3+2];
        Aw[0][p*4+3] = Jt[0*3 + p];
    }

    for (int i = 1; i < 16; ++i) {
        // euler_pose for this joint: theta . hands_comp + hands_mean
        float e[3];
        for (int c = 0; c < 3; ++c) {
            int n = 3*(i-1) + c;
            float s = hmean[n];
            for (int t = 0; t < NC; ++t) s += th[t] * hcomp[t*NC + n];
            e[c] = s;
        }
        rodr(e[0], e[1], e[2], R);
        int pr = par[i];
        float t0 = Jt[i*3+0] - Jt[pr*3+0];
        float t1 = Jt[i*3+1] - Jt[pr*3+1];
        float t2 = Jt[i*3+2] - Jt[pr*3+2];
        for (int r = 0; r < 3; ++r) {
            float a0 = Aw[pr][r*4+0], a1 = Aw[pr][r*4+1];
            float a2 = Aw[pr][r*4+2], a3 = Aw[pr][r*4+3];
            Aw[i][r*4+0] = a0*R[0] + a1*R[3] + a2*R[6];
            Aw[i][r*4+1] = a0*R[1] + a1*R[4] + a2*R[7];
            Aw[i][r*4+2] = a0*R[2] + a1*R[5] + a2*R[8];
            Aw[i][r*4+3] = a0*t0  + a1*t1  + a2*t2  + a3;
        }
    }

    // A_rel (subtract A_world . [J,0] from translation column) -> f16 A_p rows
    for (int j = 0; j < 16; ++j) {
        float jx = Jt[j*3+0], jy = Jt[j*3+1], jz = Jt[j*3+2];
        for (int p = 0; p < 3; ++p) {
            float m0 = Aw[j][p*4+0], m1 = Aw[j][p*4+1];
            float m2 = Aw[j][p*4+2], m3 = Aw[j][p*4+3];
            float t3 = m3 - (m0*jx + m1*jy + m2*jz);
            size_t base = (size_t)p*BB*64 + (size_t)b*64 + j*4;
            Ah[base+0] = (_Float16)m0;
            Ah[base+1] = (_Float16)m1;
            Ah[base+2] = (_Float16)m2;
            Ah[base+3] = (_Float16)t3;
        }
    }
}

// ---------------------------------------------------------------- kernel 3
// One wave per (btile,vtile); all three output components p fused:
// B operand (G tile) loaded once, 3x A operands, 6 chained WMMAs, and the
// per-(b,v) store is 3 contiguous floats (mergeable to global_store_b96).
__global__ void k_lbs(const unsigned short* __restrict__ Ah_u,
                      const unsigned short* __restrict__ G_u,
                      float* __restrict__ out) {
    const _Float16* Ah = reinterpret_cast<const _Float16*>(Ah_u);
    const _Float16* G  = reinterpret_cast<const _Float16*>(G_u);

    int lane = threadIdx.x & 31;
    int wid  = blockIdx.x * (blockDim.x >> 5) + (threadIdx.x >> 5);
    int vt = wid % NVT;            // 0..67
    int bt = wid / NVT;            // 0..255

    int halfsel = lane >> 4;       // K-group select (ISA operand striping)
    int idx16   = lane & 15;       // A row (M) / B column (N) within tile

    const _Float16* brow = G + (size_t)(vt*16 + idx16)*64 + halfsel*8;
    v16h b0 = load16(brow);        // K = 0..31
    v16h b1 = load16(brow + 32);   // K = 32..63

    v8f acc[3];
    #pragma unroll
    for (int p = 0; p < 3; ++p) {
        const _Float16* arow =
            Ah + (size_t)p*BB*64 + (size_t)(bt*16 + idx16)*64 + halfsel*8;
        v16h a0 = load16(arow);
        v16h a1 = load16(arow + 32);
        v8f c = {};
        c = __builtin_amdgcn_wmma_f32_16x16x32_f16(
                false, a0, false, b0, (short)0, c, false, false);
        c = __builtin_amdgcn_wmma_f32_16x16x32_f16(
                false, a1, false, b1, (short)0, c, false, false);
        acc[p] = c;
    }

    // D layout: VGPR rr -> M = rr + 8*halfsel, N = idx16
    int v = vt*16 + idx16;
    if (v < NV) {
        int M = halfsel << 3;
        #pragma unroll
        for (int rr = 0; rr < 8; ++rr) {
            float* o = out + ((size_t)(bt*16 + rr + M) * NV + v) * 3;
            o[0] = acc[0][rr];
            o[1] = acc[1][rr];
            o[2] = acc[2][rr];
        }
    }
}

// ---------------------------------------------------------------- kernel 4
__global__ void k_joints(const float* __restrict__ verts,
                         const float* __restrict__ Jreg,
                         float* __restrict__ joints) {
    int t = blockIdx.x * blockDim.x + threadIdx.x;
    if (t >= BB * 63) return;
    int b = t / 63, jc = t % 63;
    int j = jc / 3, c = jc % 3;
    const float* vb = verts + (size_t)b * NV * 3 + c;
    float acc = 0.0f;
    for (int v = 0; v < NV; ++v)
        acc += vb[v*3] * Jreg[v*NJR + j];
    joints[t] = acc;
}

// ---------------------------------------------------------------- launch
extern "C" void kernel_launch(void* const* d_in, const int* in_sizes, int n_in,
                              void* d_out, int out_size, void* d_ws, size_t ws_size,
                              hipStream_t stream) {
    (void)in_sizes; (void)n_in; (void)out_size; (void)ws_size;
    // setup_inputs order: beta(unused), theta, wrist_euler, v_template,
    //                     J_regressor, hands_comp, hands_mean, weights
    const float* theta = (const float*)d_in[1];
    const float* wrist = (const float*)d_in[2];
    const float* vtem  = (const float*)d_in[3];
    const float* Jreg  = (const float*)d_in[4];
    const float* hcomp = (const float*)d_in[5];
    const float* hmean = (const float*)d_in[6];
    const float* wgt   = (const float*)d_in[7];
    float* out = (float*)d_out;

    char* ws = (char*)d_ws;
    float*          Jt = (float*)ws;                                    // 256 B
    unsigned short* G  = (unsigned short*)(ws + 256);                   // NVP*64 f16
    unsigned short* Ah = (unsigned short*)(ws + 256 + (size_t)NVP*64*2);// 3*B*64 f16

    k_jt     <<<1, 64, 0, stream>>>(vtem, Jreg, Jt);
    k_build_G<<<(NVP*64)/256, 256, 0, stream>>>(wgt, vtem, G);
    k_pose   <<<BB/256, 256, 0, stream>>>(theta, wrist, hcomp, hmean, Jt, Ah);
    // 256 * 68 = 17408 waves, 8 waves (256 thr) per block -> 2176 blocks
    k_lbs    <<<(NBT*NVT)/8, 256, 0, stream>>>(Ah, G, out);
    k_joints <<<(BB*63)/256, 256, 0, stream>>>(out, Jreg, out + (size_t)BB*NV*3);
}
